// AOC_2954937500107
// MI455X (gfx1250) — matrix operand order repaired and checked by reading, not compiled, not run on hardware
//
#include <hip/hip_runtime.h>
#include <hip/hip_bf16.h>

typedef __attribute__((ext_vector_type(16))) _Float16 v16h;
typedef __attribute__((ext_vector_type(8)))  _Float16 v8h;
typedef __attribute__((ext_vector_type(8)))  float    v8f;

union AFrag { v16h v; v8h h[2]; };

#define BROWS 4096
#define INDIM 784
#define KPAD  800
#define HID   256
#define OUTD  10
#define TOLF  1e-3f
#define MAXIT 200

// sched_group_barrier masks
#define SGB_WMMA   0x008
#define SGB_VMEMRD 0x020
#define SGB_DSRD   0x100

// ---------------- prep kernels ----------------

__global__ void prep_misc(const float* __restrict__ b_raw,
                          const float* __restrict__ alpha_raw,
                          const float* __restrict__ beta_raw,
                          float* __restrict__ bq, float* __restrict__ scal) {
    int t = threadIdx.x;
    if (t < HID) {
        float v = rintf(tanhf(b_raw[t]) * 255.0f);
        v = fminf(fmaxf(v, -256.0f), 255.0f);
        bq[t] = v * (1.0f / 255.0f);
    }
    if (t == 0) {
        scal[0] = 1.0f / (1.0f + expf(-alpha_raw[0]));  // alpha
        scal[1] = 1.0f / (1.0f + expf(-beta_raw[0]));   // beta
    }
}

__global__ void prep_wq(const float* __restrict__ W_raw, _Float16* __restrict__ Wq) {
    int i = blockIdx.x, j = threadIdx.x;
    float w = W_raw[i * HID + j];
    float q = fminf(fmaxf(rintf(tanhf(w) * 255.0f), -256.0f), 255.0f) * (1.0f / 255.0f);
    Wq[i * HID + j] = (_Float16)q;
}

// convert float rows -> f16 rows, zero-padding columns [cols, padc)
__global__ void cvt_pad_f16(const float* __restrict__ src, _Float16* __restrict__ dst,
                            int cols, int padc) {
    int r = blockIdx.x;
    for (int c = threadIdx.x; c < padc; c += blockDim.x) {
        float v = (c < cols) ? src[(size_t)r * cols + c] : 0.0f;
        dst[(size_t)r * padc + c] = (_Float16)v;
    }
}

// ---------------- x_proj GEMM: xpb = x @ W_ip.T + b_ip + bq ----------------
// grid 64 blocks x 128 threads (4 waves); each wave: 16 rows x 256 cols, K = 800.

__global__ void __launch_bounds__(128)
xproj_kernel(const _Float16* __restrict__ xh, const _Float16* __restrict__ wip,
             const float* __restrict__ b_ip, const float* __restrict__ bq,
             float* __restrict__ xpb) {
    const int lane = threadIdx.x & 31;
    const int wave = threadIdx.x >> 5;
    const int rowBase = blockIdx.x * 64 + wave * 16;
    const int colL = lane & 15;
    const int half = lane >> 4;

    const _Float16* aBase = xh + (size_t)(rowBase + colL) * KPAD + half * 16;

#pragma unroll 1
    for (int nbh = 0; nbh < 2; ++nbh) {
        v8f acc[8];
#pragma unroll
        for (int j = 0; j < 8; ++j) acc[j] = v8f{};

        const _Float16* bBase = wip + (size_t)(nbh * 8 * 16 + colL) * KPAD + half * 16;

        for (int kb = 0; kb < KPAD / 32; ++kb) {
            AFrag a;
            a.h[0] = *(const v8h*)(aBase + kb * 32);
            a.h[1] = *(const v8h*)(aBase + kb * 32 + 8);
            v16h b[8];
#pragma unroll
            for (int j = 0; j < 8; ++j)
                b[j] = *(const v16h*)(bBase + (size_t)j * 16 * KPAD + kb * 32);
#pragma unroll
            for (int j = 0; j < 8; ++j)
                acc[j] = __builtin_amdgcn_wmma_f32_16x16x32_f16(
                    false, a.v, false, b[j], (short)0, acc[j], false, false);

            // pipeline: 6 loads ahead, then alternate 1 WMMA / 2 loads
            __builtin_amdgcn_sched_group_barrier(SGB_VMEMRD, 6, 0);
#pragma unroll
            for (int u = 0; u < 6; ++u) {
                __builtin_amdgcn_sched_group_barrier(SGB_WMMA, 1, 0);
                __builtin_amdgcn_sched_group_barrier(SGB_VMEMRD, 2, 0);
            }
            __builtin_amdgcn_sched_group_barrier(SGB_WMMA, 2, 0);
        }
#pragma unroll
        for (int j = 0; j < 8; ++j) {
            const int n = (nbh * 8 + j) * 16 + colL;
            const float addv = b_ip[n] + bq[n];
#pragma unroll
            for (int m = 0; m < 8; ++m) {
                const int row = rowBase + half * 8 + m;
                xpb[(size_t)row * HID + n] = acc[j][m] + addv;
            }
        }
    }
}

// ---------------- fixed-point iteration ----------------
// grid 64 x 128; Wq staged in LDS (128KB), per-wave tanh tile (8KB each).

#define SM_WQ   (HID * HID * 2)      // 131072 B
#define SM_TANH (16 * HID * 2)       // 8192 B per wave
#define SM_TOTAL (SM_WQ + 4 * SM_TANH)

__global__ void __launch_bounds__(128)
deq_iter_kernel(const float* __restrict__ xpb, const _Float16* __restrict__ wq,
                const float* __restrict__ scal, float* __restrict__ sOut) {
    extern __shared__ char smem[];
    _Float16* sWq = (_Float16*)smem;
    _Float16* sT  = (_Float16*)(smem + SM_WQ);
    __shared__ int flags[2];

    const int tid  = threadIdx.x;
    const int lane = tid & 31;
    const int wave = tid >> 5;
    const int rowBase = blockIdx.x * 64 + wave * 16;
    const int colL = lane & 15;
    const int half = lane >> 4;
    _Float16* myT = sT + wave * (16 * HID);

    // stage Wq (f16, row-major 256x256) into LDS: 8192 x v8h across 128 threads
    {
        const v8h* src = (const v8h*)wq;
        v8h* dst = (v8h*)sWq;
        for (int i = tid; i < (HID * HID) / 8; i += 128) dst[i] = src[i];
    }
    if (tid == 0) { flags[0] = 0; flags[1] = 0; }
    const float alpha = scal[0];
    const float beta  = scal[1];
    __syncthreads();

    // per-lane base pointers (constant-fold inner offsets)
    const float* xpbBase = xpb + (size_t)(rowBase + half * 8) * HID + colL;
    float* sOutBase = sOut + (size_t)(rowBase + half * 8) * HID + colL;
    _Float16* tWr = myT + (half * 8) * HID + colL;           // tanh writes
    const _Float16* tRd = myT + colL * HID + half * 16;      // A-frag reads
    const _Float16* bRd = sWq + (size_t)colL * HID + half * 16;

    // s0 = xpb, held in registers in WMMA C/D layout
    v8f s[16];
#pragma unroll
    for (int nb = 0; nb < 16; ++nb)
#pragma unroll
        for (int m = 0; m < 8; ++m)
            s[nb][m] = xpbBase[m * HID + nb * 16];

    unsigned convMask = 0;  // sticky per-row convergence, rows half*8+m
    for (int it = 0; it < MAXIT; ++it) {
        // tanh(s) -> per-wave LDS tile, f16 row-major [16][256]
#pragma unroll
        for (int nb = 0; nb < 16; ++nb)
#pragma unroll
            for (int m = 0; m < 8; ++m) {
                const float xv = s[nb][m];
                const float e = __expf(2.0f * xv);
                const float t = 1.0f - 2.0f / (e + 1.0f);
                tWr[m * HID + nb * 16] = (_Float16)t;
            }

        // preload all 8 A fragments (per-wave LDS; DS ops are in-order per wave)
        AFrag a[8];
#pragma unroll
        for (int kb = 0; kb < 8; ++kb) {
            a[kb].h[0] = *(const v8h*)(tRd + kb * 32);
            a[kb].h[1] = *(const v8h*)(tRd + kb * 32 + 8);
        }

        float dmax[8];
#pragma unroll
        for (int m = 0; m < 8; ++m) dmax[m] = 0.0f;

#pragma unroll 1
        for (int nb = 0; nb < 16; ++nb) {
            const _Float16* bp = bRd + (size_t)nb * 16 * HID;

            // prefetch epilogue operands (L2) so they land behind the WMMA chain
            float xv[8];
#pragma unroll
            for (int m = 0; m < 8; ++m) xv[m] = xpbBase[m * HID + nb * 16];

            v16h b[8];
#pragma unroll
            for (int kb = 0; kb < 8; ++kb)
                b[kb] = *(const v16h*)(bp + kb * 32);

            v8f g = v8f{};
#pragma unroll
            for (int kb = 0; kb < 8; ++kb)
                g = __builtin_amdgcn_wmma_f32_16x16x32_f16(
                    false, a[kb].v, false, b[kb], (short)0, g, false, false);

#pragma unroll
            for (int m = 0; m < 8; ++m) {
                const float sn = alpha * s[nb][m] + beta * g[m] + xv[m];
                dmax[m] = fmaxf(dmax[m], fabsf(sn - s[nb][m]));
                s[nb][m] = sn;
            }

            // pipeline: xpb loads first, 3 B frags ahead, then 1 WMMA / 1 frag
            __builtin_amdgcn_sched_group_barrier(SGB_VMEMRD, 8, 0);
            __builtin_amdgcn_sched_group_barrier(SGB_DSRD, 6, 0);
#pragma unroll
            for (int u = 0; u < 5; ++u) {
                __builtin_amdgcn_sched_group_barrier(SGB_WMMA, 1, 0);
                __builtin_amdgcn_sched_group_barrier(SGB_DSRD, 2, 0);
            }
            __builtin_amdgcn_sched_group_barrier(SGB_WMMA, 3, 0);
        }

        // per-row inf-norm: reduce across the 16 lanes of each half-wave
#pragma unroll
        for (int m = 0; m < 8; ++m) {
            float d = dmax[m];
            d = fmaxf(d, __shfl_xor(d, 1, 32));
            d = fmaxf(d, __shfl_xor(d, 2, 32));
            d = fmaxf(d, __shfl_xor(d, 4, 32));
            d = fmaxf(d, __shfl_xor(d, 8, 32));
            if (d < TOLF) convMask |= (1u << m);
        }
        const int nd = (convMask != 0xffu) ? 1 : 0;
        const bool anynd = (__ballot(nd) != 0ull);
        if (lane == 0 && anynd) atomicOr(&flags[it & 1], 1);
        __syncthreads();                       // all votes for slot it&1 landed
        const int f = flags[it & 1];
        if (tid == 0) flags[(it + 1) & 1] = 0; // prep other slot
        __syncthreads();                       // reads done + reset visible
        if (f == 0) break;                     // uniform across block
    }

    // write final s
#pragma unroll
    for (int nb = 0; nb < 16; ++nb)
#pragma unroll
        for (int m = 0; m < 8; ++m)
            sOutBase[m * HID + nb * 16] = s[nb][m];
}

// ---------------- output GEMM (tiny, f32 for accuracy) ----------------

__global__ void out_kernel(const float* __restrict__ s, const float* __restrict__ W_op,
                           const float* __restrict__ b_op, float* __restrict__ out) {
    const int idx = blockIdx.x * blockDim.x + threadIdx.x;
    if (idx >= BROWS * OUTD) return;
    const int row = idx / OUTD, o = idx % OUTD;
    const float* sr = s + (size_t)row * HID;
    const float* wr = W_op + (size_t)o * HID;
    float acc = 0.0f;
#pragma unroll 8
    for (int k = 0; k < HID; ++k) acc = fmaf(sr[k], wr[k], acc);
    out[idx] = acc + b_op[o];
}

// ---------------- launcher ----------------

#define OFF_XH    ((size_t)0)                          // 4096*800*2 = 6,553,600
#define OFF_WIP   ((size_t)6553600)                    // 256*800*2  =   409,600
#define OFF_WQ    ((size_t)6963200)                    // 256*256*2  =   131,072
#define OFF_XPB   ((size_t)7094272)                    // 4096*256*4 = 4,194,304
#define OFF_S     ((size_t)11288576)                   // 4096*256*4 = 4,194,304
#define OFF_BQ    ((size_t)15482880)                   // 256*4
#define OFF_SCAL  ((size_t)15483904)                   // 2*4

extern "C" void kernel_launch(void* const* d_in, const int* in_sizes, int n_in,
                              void* d_out, int out_size, void* d_ws, size_t ws_size,
                              hipStream_t stream) {
    const float* x      = (const float*)d_in[0];
    const float* W_ip   = (const float*)d_in[1];
    const float* b_ip   = (const float*)d_in[2];
    const float* W_op   = (const float*)d_in[3];
    const float* b_op   = (const float*)d_in[4];
    const float* W_raw  = (const float*)d_in[5];
    const float* b_raw  = (const float*)d_in[6];
    const float* a_raw  = (const float*)d_in[7];
    const float* be_raw = (const float*)d_in[8];

    char* ws = (char*)d_ws;
    _Float16* xh   = (_Float16*)(ws + OFF_XH);
    _Float16* wip  = (_Float16*)(ws + OFF_WIP);
    _Float16* wqh  = (_Float16*)(ws + OFF_WQ);
    float*    xpb  = (float*)(ws + OFF_XPB);
    float*    sbuf = (float*)(ws + OFF_S);
    float*    bq   = (float*)(ws + OFF_BQ);
    float*    scal = (float*)(ws + OFF_SCAL);
    float*    out  = (float*)d_out;

    prep_misc<<<1, 256, 0, stream>>>(b_raw, a_raw, be_raw, bq, scal);
    prep_wq<<<HID, HID, 0, stream>>>(W_raw, wqh);
    cvt_pad_f16<<<HID, 256, 0, stream>>>(W_ip, wip, INDIM, KPAD);
    cvt_pad_f16<<<BROWS, 256, 0, stream>>>(x, xh, INDIM, KPAD);

    xproj_kernel<<<BROWS / 64, 128, 0, stream>>>(xh, wip, b_ip, bq, xpb);
    deq_iter_kernel<<<BROWS / 64, 128, SM_TOTAL, stream>>>(xpb, wqh, scal, sbuf);
    out_kernel<<<(BROWS * OUTD + 255) / 256, 256, 0, stream>>>(sbuf, W_op, b_op, out);
}
